// CSOCSSC_v113_91207925498231
// MI455X (gfx1250) — matrix operand order repaired and checked by reading, not compiled, not run on hardware
//
#include <hip/hip_runtime.h>
#include <hip/hip_bf16.h>
#include <cstdint>
#include <cstddef>

// Problem constants (match reference)
#define D_MODEL 1024
#define NSEQ    4096
#define NBATCH  2
#define EPS_K   1e-4f

typedef __attribute__((ext_vector_type(16))) _Float16 v16h;
typedef __attribute__((ext_vector_type(8)))  float    v8f;
typedef __attribute__((ext_vector_type(4)))  int      v4i;

union FragU { uint4 u[2]; v16h h; };

// ---------------------------------------------------------------------------
// CDNA5 async global->LDS (GLOBAL_LOAD_ASYNC_TO_LDS_B128, ASYNCcnt-tracked).
// Signature (from compiler diagnostic): (AS1 v4i*, AS3 v4i*, imm off, imm cpol)
// Guarded: falls back to a synchronous register-path copy if absent.
// ---------------------------------------------------------------------------
#if __has_builtin(__builtin_amdgcn_global_load_async_to_lds_b128)
#define HAVE_ASYNC_LDS 1
#endif

__device__ __forceinline__ void async_copy_b128(void* lds_dst, const void* gsrc) {
#ifdef HAVE_ASYNC_LDS
  __builtin_amdgcn_global_load_async_to_lds_b128(
      (__attribute__((address_space(1))) v4i*)gsrc,
      (__attribute__((address_space(3))) v4i*)lds_dst,
      /*imm offset=*/0, /*cpol=*/0);
#else
  *(uint4*)lds_dst = *(const uint4*)gsrc;
#endif
}

__device__ __forceinline__ void async_fence() {
#ifdef HAVE_ASYNC_LDS
#if __has_builtin(__builtin_amdgcn_s_wait_asynccnt)
  __builtin_amdgcn_s_wait_asynccnt(0);
#else
  asm volatile("s_wait_asynccnt 0x0" ::: "memory");
#endif
#endif
}

__device__ __forceinline__ v8f wmma16(v16h a, v16h b, v8f c) {
  // D = A(16x32 f16) * B(32x16 f16) + C(16x16 f32)
  return __builtin_amdgcn_wmma_f32_16x16x32_f16(
      /*neg_a=*/false, a, /*neg_b=*/false, b,
      /*c_mod=*/(short)0, c, /*reuse_a=*/false, /*reuse_b=*/false);
}

// A-matrix fragment 16x32 f16 from LDS tile stored row-major [M][K],
// stride in halfs, stride % 8 == 0, base 16B-aligned.
// ISA layout: lanes 0-15 (M=r): elems 0..7 -> K=0..7, 8..15 -> K=16..23
//             lanes 16-31 (M=r): elems 0..7 -> K=8..15, 8..15 -> K=24..31
__device__ __forceinline__ v16h frag_a(const _Float16* base, int stride, int lane) {
  const int r = lane & 15, hh = (lane >> 4) & 1;
  const _Float16* p = base + (size_t)r * stride;
  FragU f;
  f.u[0] = *(const uint4*)(p + 8 * hh);
  f.u[1] = *(const uint4*)(p + 16 + 8 * hh);
  return f.h;
}

// B-matrix fragment 32x16 f16; LDS tile stored as [N rows][K cols]; per-lane
// read N = lane%16, elems 0..7 -> K = 8*half + e, 8..15 -> K = 16 + 8*half + (e-8).
__device__ __forceinline__ v16h frag_b(const _Float16* base, int stride, int lane) {
  const int r = lane & 15, hh = (lane >> 4) & 1;
  const _Float16* p = base + (size_t)r * stride;
  FragU f;
  f.u[0] = *(const uint4*)(p + 8 * hh);
  f.u[1] = *(const uint4*)(p + 16 + 8 * hh);
  return f.h;
}

// ---------------------------------------------------------------------------
// Generic WMMA GEMM: C[m,n] = sum_k A[m,k] * B(k,n) (+ bias[n])
//   TRANSPOSE_B == false: Bm is [n][k] row-major (ldb along k)   (x @ W^T)
//   TRANSPOSE_B == true : Bm is [k][n] row-major (ldb along n)   (attn @ v)
// Block tile 128x128, BK=32, 256 threads = 8 waves in a 4(M) x 2(N) grid.
// f16 staging goes through the CDNA5 async global->LDS path.
// ---------------------------------------------------------------------------
template <typename AT, typename BT, bool TRANSPOSE_B, bool HAS_BIAS, typename OT>
__global__ __launch_bounds__(256)
void gemm_wmma_kernel(const AT* __restrict__ A, int lda,
                      const BT* __restrict__ Bm, int ldb,
                      const float* __restrict__ bias,
                      OT* __restrict__ C, int ldc, int Ktot) {
  __shared__ __align__(16) _Float16 As[128 * 40];
  __shared__ __align__(16) _Float16 Bs[128 * 40];

  const int tid  = threadIdx.x;
  const int lane = tid & 31;
  const int wave = tid >> 5;
  const int wm   = wave >> 1;  // 0..3 (M group of 32 rows)
  const int wn   = wave & 1;   // 0..1 (N group of 64 cols)
  const int m0   = blockIdx.y * 128;
  const int n0   = blockIdx.x * 128;

  const v8f vzero = {0.f, 0.f, 0.f, 0.f, 0.f, 0.f, 0.f, 0.f};
  v8f acc[2][4];
#pragma unroll
  for (int i = 0; i < 2; ++i)
#pragma unroll
    for (int j = 0; j < 4; ++j) acc[i][j] = vzero;

  for (int k0 = 0; k0 < Ktot; k0 += 32) {
    {  // stage A tile: 128 rows x 32 K
      const int row = tid >> 1, cb = (tid & 1) << 4;
      const AT* g = A + (size_t)(m0 + row) * lda + (k0 + cb);
      if constexpr (sizeof(AT) == 4) {  // f32 -> f16 convert via registers
        _Float16 t16[16];
#pragma unroll
        for (int i = 0; i < 16; i += 4) {
          float4 f = *(const float4*)(g + i);
          t16[i + 0] = (_Float16)f.x; t16[i + 1] = (_Float16)f.y;
          t16[i + 2] = (_Float16)f.z; t16[i + 3] = (_Float16)f.w;
        }
        *(uint4*)(As + row * 40 + cb)     = *(const uint4*)(t16);
        *(uint4*)(As + row * 40 + cb + 8) = *(const uint4*)(t16 + 8);
      } else {  // f16: async DMA straight into LDS
        async_copy_b128(As + row * 40 + cb,     g);
        async_copy_b128(As + row * 40 + cb + 8, g + 8);
      }
      if (k0 + 32 < Ktot) __builtin_prefetch(g + 32, 0, 3);  // global_prefetch_b8
    }
    if constexpr (!TRANSPOSE_B) {  // Bm[n][k] -> direct copy into [N][K]
      const int row = tid >> 1, cb = (tid & 1) << 4;
      const BT* g = Bm + (size_t)(n0 + row) * ldb + (k0 + cb);
      if constexpr (sizeof(BT) == 4) {
        _Float16 t16[16];
#pragma unroll
        for (int i = 0; i < 16; i += 4) {
          float4 f = *(const float4*)(g + i);
          t16[i + 0] = (_Float16)f.x; t16[i + 1] = (_Float16)f.y;
          t16[i + 2] = (_Float16)f.z; t16[i + 3] = (_Float16)f.w;
        }
        *(uint4*)(Bs + row * 40 + cb)     = *(const uint4*)(t16);
        *(uint4*)(Bs + row * 40 + cb + 8) = *(const uint4*)(t16 + 8);
      } else {
        async_copy_b128(Bs + row * 40 + cb,     g);
        async_copy_b128(Bs + row * 40 + cb + 8, g + 8);
      }
      if (k0 + 32 < Ktot) __builtin_prefetch(g + 32, 0, 3);
    } else {  // Bm[k][n] (f16) -> transpose into [N][K] during staging
      const int kk = tid >> 3, nb = (tid & 7) << 4;
      const BT* g = Bm + (size_t)(k0 + kk) * ldb + (n0 + nb);
      _Float16 t16[16];
      *(uint4*)(t16)     = *(const uint4*)(g);
      *(uint4*)(t16 + 8) = *(const uint4*)(g + 8);
#pragma unroll
      for (int i = 0; i < 16; ++i) Bs[(size_t)(nb + i) * 40 + kk] = t16[i];
      if (k0 + 32 < Ktot) __builtin_prefetch(g + (size_t)32 * ldb, 0, 3);
    }
    async_fence();
    __syncthreads();

    const v16h a0 = frag_a(As + (size_t)(wm * 32 + 0) * 40, 40, lane);
    const v16h a1 = frag_a(As + (size_t)(wm * 32 + 16) * 40, 40, lane);
#pragma unroll
    for (int nj = 0; nj < 4; ++nj) {
      const v16h bf = frag_b(Bs + (size_t)(wn * 64 + nj * 16) * 40, 40, lane);
      acc[0][nj] = wmma16(a0, bf, acc[0][nj]);
      acc[1][nj] = wmma16(a1, bf, acc[1][nj]);
    }
    __syncthreads();
  }

  // Epilogue: C/D layout -> M = j + 8*half, N = lane%16
  const int r = lane & 15, hh = (lane >> 4) & 1;
#pragma unroll
  for (int mi = 0; mi < 2; ++mi)
#pragma unroll
    for (int nj = 0; nj < 4; ++nj) {
      const int col = n0 + wn * 64 + nj * 16 + r;
      float bv = 0.f;
      if constexpr (HAS_BIAS) bv = bias[col];
#pragma unroll
      for (int j = 0; j < 8; ++j) {
        const int row = m0 + wm * 32 + mi * 16 + hh * 8 + j;
        C[(size_t)row * ldc + col] = (OT)(acc[mi][nj][j] + bv);
      }
    }
}

// ---------------------------------------------------------------------------
// K(r) = (r+eps)^(-|alpha|) * exp(-r/|lambda|)
// ---------------------------------------------------------------------------
__global__ __launch_bounds__(256)
void kmat_kernel(float* __restrict__ Kout, const float* __restrict__ alpha_p,
                 const float* __restrict__ lam_p) {
  const float aAbs  = fabsf(alpha_p[0]);
  const float invLm = 1.f / fabsf(lam_p[0]);
  const int j = blockIdx.x * 256 + threadIdx.x;
  const int i = blockIdx.y;
  const float r = fabsf((float)(i - j));
  Kout[(size_t)i * NSEQ + j] = expf(-aAbs * logf(r + EPS_K) - r * invLm);
}

__device__ __forceinline__ float kfac(float aAbs, float invLm, float r) {
  return expf(-aAbs * logf(r + EPS_K) - r * invLm);
}

// ---------------------------------------------------------------------------
// Attention scores: per block 32 queries (LDS-resident Q tile), full 4096-key
// row. Pass 0: WMMA q*k^T, scale by 1/sqrt(d) * K(r), row-max.
// Pass 1: recompute S, p = exp(s - max), write unnormalized p, row-sums.
// All f16 staging uses the async global->LDS path.
// ---------------------------------------------------------------------------
__global__ __launch_bounds__(256)
void attn_scores_kernel(const _Float16* __restrict__ q16,
                        const _Float16* __restrict__ k16,
                        float* __restrict__ attn,         // [B,N,N] (unnormalized exp)
                        float* __restrict__ rowsum_out,   // [B*N]
                        const float* __restrict__ alpha_p,
                        const float* __restrict__ lam_p) {
  __shared__ __align__(16) _Float16 Qs[32 * 1032];  // 32 q rows x 1024 d (+pad)
  __shared__ __align__(16) _Float16 Bs[256 * 40];   // 256 keys x 32 d (+pad)
  __shared__ float pred[32][4];
  __shared__ float rowmax[32];
  __shared__ float rowsum[32];

  const int tid = threadIdx.x, lane = tid & 31;
  const int wave = tid >> 5;
  const int wq = wave >> 2;  // 0..1 -> 16-query group
  const int wk = wave & 3;   // 0..3 -> 64-key group
  const int b  = blockIdx.y;
  const int q0 = blockIdx.x * 32;
  const int r = lane & 15, hh = (lane >> 4) & 1;

  const float aAbs  = fabsf(alpha_p[0]);
  const float invLm = 1.f / fabsf(lam_p[0]);
  const float scale = 0.03125f;  // 1/sqrt(1024)

  {  // load Q tile once (async DMA, 64KB region reused for all 4096 keys)
    const int row = tid >> 3;        // 0..31
    const int cs  = (tid & 7) << 7;  // 0,128,...,896
    const _Float16* g = q16 + ((size_t)(b * NSEQ + q0 + row)) * D_MODEL + cs;
#pragma unroll
    for (int i = 0; i < 16; ++i)
      async_copy_b128(Qs + (size_t)row * 1032 + cs + i * 8, g + i * 8);
  }
  if (tid < 32) { rowmax[tid] = -3.0e38f; rowsum[tid] = 0.f; }
  async_fence();
  __syncthreads();

  const v8f vzero = {0.f, 0.f, 0.f, 0.f, 0.f, 0.f, 0.f, 0.f};

  for (int pass = 0; pass < 2; ++pass) {
    for (int kb = 0; kb < NSEQ; kb += 256) {
      v8f acc[4];
#pragma unroll
      for (int nj = 0; nj < 4; ++nj) acc[nj] = vzero;

      for (int d0 = 0; d0 < D_MODEL; d0 += 32) {
        {  // stage 256 keys x 32 dims (thread = one key row, async DMA)
          const _Float16* g = k16 + ((size_t)(b * NSEQ + kb + tid)) * D_MODEL + d0;
#pragma unroll
          for (int i = 0; i < 4; ++i)
            async_copy_b128(Bs + (size_t)tid * 40 + i * 8, g + i * 8);
          if (d0 + 32 < D_MODEL) __builtin_prefetch(g + 32, 0, 3);
        }
        async_fence();
        __syncthreads();
        const v16h a0 = frag_a(Qs + (size_t)(wq * 16) * 1032 + d0, 1032, lane);
#pragma unroll
        for (int nj = 0; nj < 4; ++nj) {
          const v16h bf = frag_b(Bs + (size_t)(wk * 64 + nj * 16) * 40, 40, lane);
          acc[nj] = wmma16(a0, bf, acc[nj]);
        }
        __syncthreads();
      }

      // scale * distance-kernel, then row reduction
      float red[8];
#pragma unroll
      for (int j = 0; j < 8; ++j) red[j] = (pass == 0) ? -3.0e38f : 0.f;
#pragma unroll
      for (int nj = 0; nj < 4; ++nj) {
        const int kidx = kb + wk * 64 + nj * 16 + r;
#pragma unroll
        for (int j = 0; j < 8; ++j) {
          const int rowl = wq * 16 + hh * 8 + j;
          const int qidx = q0 + rowl;
          const float rd = fabsf((float)(qidx - kidx));
          const float s  = acc[nj][j] * scale * kfac(aAbs, invLm, rd);
          if (pass == 0) {
            red[j] = fmaxf(red[j], s);
          } else {
            const float p = expf(s - rowmax[rowl]);
            red[j] += p;
            attn[((size_t)b * NSEQ + qidx) * NSEQ + kidx] = p;
          }
        }
      }
#pragma unroll
      for (int j = 0; j < 8; ++j) {  // reduce across the 16-lane group
#pragma unroll
        for (int m = 8; m >= 1; m >>= 1) {
          const float o = __shfl_xor(red[j], m, 32);
          red[j] = (pass == 0) ? fmaxf(red[j], o) : (red[j] + o);
        }
      }
      if (r == 0) {
#pragma unroll
        for (int j = 0; j < 8; ++j) pred[wq * 16 + hh * 8 + j][wk] = red[j];
      }
      __syncthreads();
      if (tid < 32) {
        const float v0 = pred[tid][0], v1 = pred[tid][1];
        const float v2 = pred[tid][2], v3 = pred[tid][3];
        if (pass == 0)
          rowmax[tid] = fmaxf(rowmax[tid], fmaxf(fmaxf(v0, v1), fmaxf(v2, v3)));
        else
          rowsum[tid] += (v0 + v1) + (v2 + v3);
      }
      __syncthreads();
    }
  }
  if (tid < 32) rowsum_out[b * NSEQ + q0 + tid] = rowsum[tid];
}

// ---------------------------------------------------------------------------
// Normalize attn in place and emit an f16 copy for the attn @ v GEMM.
// ---------------------------------------------------------------------------
__global__ __launch_bounds__(256)
void attn_norm_kernel(float* __restrict__ attn, _Float16* __restrict__ attn16,
                      const float* __restrict__ rowsum) {
  const int col = blockIdx.x * 256 + threadIdx.x;
  const int row = blockIdx.y;
  const int b   = blockIdx.z;
  const float inv = 1.f / rowsum[b * NSEQ + row];
  const size_t idx = ((size_t)b * NSEQ + row) * NSEQ + col;
  const float v = attn[idx] * inv;
  attn[idx]   = v;
  attn16[idx] = (_Float16)v;
}

// ---------------------------------------------------------------------------
extern "C" void kernel_launch(void* const* d_in, const int* in_sizes, int n_in,
                              void* d_out, int out_size, void* d_ws, size_t ws_size,
                              hipStream_t stream) {
  const float* x     = (const float*)d_in[0];
  const float* Wq    = (const float*)d_in[1];
  const float* bq    = (const float*)d_in[2];
  const float* Wk    = (const float*)d_in[3];
  const float* bk    = (const float*)d_in[4];
  const float* Wv    = (const float*)d_in[5];
  const float* bv    = (const float*)d_in[6];
  const float* alpha = (const float*)d_in[7];
  const float* lam   = (const float*)d_in[8];

  // d_out = concat(out[B,N,D], attn[B,N,N], K[N,N]) as f32
  float* out_o  = (float*)d_out;
  float* attn_o = out_o + (size_t)NBATCH * NSEQ * D_MODEL;
  float* kmat_o = attn_o + (size_t)NBATCH * NSEQ * NSEQ;

  // Workspace layout (~113 MiB): q16/k16/v16 (f16), attn16 (f16), rowsum (f32)
  _Float16* q16 = (_Float16*)d_ws;
  _Float16* k16 = q16 + (size_t)NBATCH * NSEQ * D_MODEL;
  _Float16* v16 = k16 + (size_t)NBATCH * NSEQ * D_MODEL;
  _Float16* a16 = v16 + (size_t)NBATCH * NSEQ * D_MODEL;
  float* rowsum = (float*)(a16 + (size_t)NBATCH * NSEQ * NSEQ);

  const dim3 blk(256);

  // 1) QKV projections: [8192,1024] x [1024,1024]^T (+bias) -> f16
  const dim3 g1(D_MODEL / 128, (NBATCH * NSEQ) / 128);
  gemm_wmma_kernel<float, float, false, true, _Float16>
      <<<g1, blk, 0, stream>>>(x, D_MODEL, Wq, D_MODEL, bq, q16, D_MODEL, D_MODEL);
  gemm_wmma_kernel<float, float, false, true, _Float16>
      <<<g1, blk, 0, stream>>>(x, D_MODEL, Wk, D_MODEL, bk, k16, D_MODEL, D_MODEL);
  gemm_wmma_kernel<float, float, false, true, _Float16>
      <<<g1, blk, 0, stream>>>(x, D_MODEL, Wv, D_MODEL, bv, v16, D_MODEL, D_MODEL);

  // 2) Distance kernel matrix K[N,N]
  kmat_kernel<<<dim3(NSEQ / 256, NSEQ), blk, 0, stream>>>(kmat_o, alpha, lam);

  // 3) Scores + softmax statistics (unnormalized exp into attn region)
  attn_scores_kernel<<<dim3(NSEQ / 32, NBATCH), blk, 0, stream>>>(
      q16, k16, attn_o, rowsum, alpha, lam);

  // 4) Normalize attn (f32 output) + f16 copy for the PV GEMM
  attn_norm_kernel<<<dim3(NSEQ / 256, NSEQ, NBATCH), blk, 0, stream>>>(
      attn_o, a16, rowsum);

  // 5) out = attn @ v per batch: [4096,4096] x [4096,1024] -> f32
  const dim3 g4(D_MODEL / 128, NSEQ / 128);
  for (int b = 0; b < NBATCH; ++b) {
    gemm_wmma_kernel<_Float16, _Float16, true, false, float>
        <<<g4, blk, 0, stream>>>(a16 + (size_t)b * NSEQ * NSEQ, NSEQ,
                                 v16 + (size_t)b * NSEQ * D_MODEL, D_MODEL,
                                 nullptr,
                                 out_o + (size_t)b * NSEQ * D_MODEL, D_MODEL, NSEQ);
  }
}
